// BiLSTM_CRF_71502615543983
// MI455X (gfx1250) — compile-verified, matrix operand order checked by reading, not run on hardware
//
#include <hip/hip_runtime.h>
#include <hip/hip_bf16.h>
#include <math.h>
#include <cstdint>

// ---------------------------------------------------------------------------
// BiLSTM-CRF for MI455X (gfx1250, wave32, WMMA + TDM).
//   1) embed gather -> bf16 X
//   2) weight conversions (bf16) + Whh transpose->packed-bf16
//   3) Zx = X @ Wih^T + b  -- TDM tensor_load_to_lds staging + v_wmma bf16
//   4) sequential LSTM recurrence, 1 WGP/direction, h in LDS, bf16 Whh stream
//   5) feats = [hf|hb] @ W_out^T + b_out  -- WMMA
//   6) Viterbi forward + backtrack, single workgroup
// ---------------------------------------------------------------------------

typedef __bf16 bf16_t;
typedef __attribute__((ext_vector_type(16))) __bf16    v16bf;
typedef __attribute__((ext_vector_type(8)))  float     v8f;
typedef __attribute__((ext_vector_type(4)))  uint32_t  u32x4;
typedef __attribute__((ext_vector_type(8)))  uint32_t  u32x8;

#define S_LEN  2048
#define E_DIM  256
#define HD_DIM 256
#define G4     1024   // 4*HD
#define H_DIM  512
#define T_TAGS 48

// workspace layout (bytes)
#define OFF_XBF    ((size_t)0x00000000)  // S*E bf16            = 1 MB
#define OFF_WIHF   ((size_t)0x00100000)  // 1024*256 bf16       = 512 KB
#define OFF_WIHB   ((size_t)0x00180000)  // 512 KB
#define OFF_WOUTBF ((size_t)0x00200000)  // 48*512 bf16 (padded)
#define OFF_WHHTF  ((size_t)0x00210000)  // 128*1024 u32 (packed bf16 pairs) = 512 KB
#define OFF_WHHTB  ((size_t)0x00310000)  // 512 KB
#define OFF_ZXF    ((size_t)0x00410000)  // 2048*1024 f32       = 8 MB
#define OFF_ZXB    ((size_t)0x00C10000)  // 8 MB
#define OFF_HF     ((size_t)0x01410000)  // 2048*256 f32        = 2 MB
#define OFF_HB     ((size_t)0x01610000)  // 2 MB
#define OFF_FEATS  ((size_t)0x01810000)  // 2048*48 f32 (padded)
#define OFF_BPS    ((size_t)0x01880000)  // 2047*48 i32

// ISA 7.12.2: 16-bit A/B fragment per-lane K map (8 VGPRs x 2 packed).
__device__ __forceinline__ int wmma_kmap(int e, int hi) {
    return ((e < 8) ? 0 : 16) + (e & 7) + (hi ? 8 : 0);
}

// ---------------------------------------------------------------------------
// TDM: DMA one contiguous 8KB block (2048 dword elements) from global to LDS.
// D# per CDNA5 ISA: group0 (4 SGPRs) + group1 (8 SGPRs),
// 2D descriptor with tile_dim0 = 2048 x tile_dim1 = 1, data_size = 4B.
// ---------------------------------------------------------------------------
__device__ __forceinline__ void tdm_load_8kb(uint32_t lds_addr, const void* gptr) {
    const uint64_t ga = (uint64_t)(uintptr_t)gptr;
    u32x4 g0;
    g0.x = 1u;                                     // count=1, user mode, no gather
    g0.y = lds_addr;                               // LDS byte address
    g0.z = (uint32_t)ga;                           // global_addr[31:0]
    g0.w = (uint32_t)((ga >> 32) & 0x01FFFFFFu)    // global_addr[56:32]
         | (2u << 30);                             // type = 2 ("image")
    u32x8 g1;
    g1[0] = (2u << 16);                            // data_size=4B, wg_mask=0
    g1[1] = (2048u & 0xFFFFu) << 16;               // tensor_dim0[15:0]
    g1[2] = (2048u >> 16) | (1u << 16);            // tensor_dim0[31:16] | tensor_dim1[15:0]=1
    g1[3] = (2048u << 16);                         // tensor_dim1[31:16]=0 | tile_dim0=2048
    g1[4] = 1u;                                    // tile_dim1=1, tile_dim2=0
    g1[5] = 2048u;                                 // tensor_dim0_stride[31:0]
    g1[6] = 0u;                                    // stride0 hi | stride1 lo
    g1[7] = 0u;                                    // stride1 hi
    asm volatile("tensor_load_to_lds %0, %1" :: "s"(g0), "s"(g1) : "memory");
}

// LDS byte offset of a shared-memory pointer (ISA §10.2: LDS aperture maps
// flat addr[31:0] to the LDS offset).
__device__ __forceinline__ uint32_t lds_off(const void* p) {
    return (uint32_t)(uintptr_t)p;
}

// ---------------- 1) embedding gather -> bf16 ------------------------------
__global__ void k_embed(const int* __restrict__ sent,
                        const float* __restrict__ table,
                        bf16_t* __restrict__ xbf) {
    const int s = blockIdx.x;
    const long row = (long)sent[s];
    const float* src = table + row * E_DIM;
    bf16_t* dst = xbf + (long)s * E_DIM;
    for (int e = threadIdx.x; e < E_DIM; e += blockDim.x)
        dst[e] = (bf16_t)src[e];
}

// ---------------- 2a) generic f32 -> bf16 ----------------------------------
__global__ void k_cvt_bf16(const float* __restrict__ src,
                           bf16_t* __restrict__ dst, int n) {
    int i = blockIdx.x * blockDim.x + threadIdx.x;
    if (i < n) dst[i] = (bf16_t)src[i];
}

// ------- 2b) transpose Whh (G4 x HD) f32 -> (HD/2 x G4) packed bf16 --------
__global__ void k_transpose_pack(const float* __restrict__ src,
                                 uint32_t* __restrict__ dst) {
    int i = blockIdx.x * blockDim.x + threadIdx.x;   // over (HD/2)*G4 outputs
    if (i < (HD_DIM / 2) * G4) {
        const int k2 = i / G4;          // 0..127
        const int j  = i - k2 * G4;     // 0..1023 (gate row)
        union { bf16_t h[2]; uint32_t u; } p;
        p.h[0] = (bf16_t)src[j * HD_DIM + 2 * k2];
        p.h[1] = (bf16_t)src[j * HD_DIM + 2 * k2 + 1];
        dst[i] = p.u;
    }
}

// ---------------- 3) Zx = X @ Wih^T + b  (both directions) -----------------
// grid (128 Mtiles, 64 Ntiles, 2 dirs), block = 32 (one wave per 16x16 tile).
// X and W tiles (each 16 contiguous rows = 8KB) staged to LDS by the TDM.
__global__ void k_gemm_zx(const bf16_t* __restrict__ xbf,
                          const bf16_t* __restrict__ wf,
                          const bf16_t* __restrict__ wb,
                          const float* __restrict__ biasf,
                          const float* __restrict__ biasb,
                          float* __restrict__ zxf,
                          float* __restrict__ zxb) {
    __shared__ __align__(16) bf16_t sx[16 * E_DIM];   // 8 KB
    __shared__ __align__(16) bf16_t sw[16 * E_DIM];   // 8 KB

    const int lane = threadIdx.x;
    const int mn   = lane & 15;
    const int hi   = lane >> 4;
    const int mt   = blockIdx.x;
    const int nt   = blockIdx.y;
    const int dir  = blockIdx.z;

    const bf16_t* W    = dir ? wb : wf;
    const float*  bias = dir ? biasb : biasf;
    float*        Z    = dir ? zxb : zxf;

    // TDM DMA both tiles into LDS, then wait on TENSORcnt.
    tdm_load_8kb(lds_off(sx), xbf + (size_t)mt * 16 * E_DIM);
    tdm_load_8kb(lds_off(sw), W   + (size_t)nt * 16 * E_DIM);
    __builtin_amdgcn_s_wait_tensorcnt(0);

    const bf16_t* arow = sx + mn * E_DIM;
    const bf16_t* brow = sw + mn * E_DIM;

    union { v8f v; float s[8]; } acc;
    acc.v = (v8f){0.f,0.f,0.f,0.f,0.f,0.f,0.f,0.f};

    for (int k0 = 0; k0 < E_DIM; k0 += 32) {
        union { v16bf v; bf16_t s[16]; } a, b;
#pragma unroll
        for (int e = 0; e < 16; ++e) {
            const int k = k0 + wmma_kmap(e, hi);   // two contiguous 8-elem runs -> ds_load_b128
            a.s[e] = arow[k];
            b.s[e] = brow[k];
        }
        acc.v = __builtin_amdgcn_wmma_f32_16x16x32_bf16(
            false, a.v, false, b.v, (short)0, acc.v, false, false);
    }

    const int col = nt * 16 + mn;
    const float bc = bias[col];
#pragma unroll
    for (int r = 0; r < 8; ++r) {
        const int row = mt * 16 + r + hi * 8;
        Z[(long)row * G4 + col] = acc.s[r] + bc;
    }
}

// ---------------- 4) sequential LSTM recurrence ----------------------------
// grid = 2 (dir), block = 1024 threads (32 waves on one WGP).
// Whh^T streamed as packed bf16 pairs (halves per-step L2 traffic);
// unpack is shift/mask only. h broadcast from LDS as float2.
__global__ __launch_bounds__(1024)
void k_lstm(const float* __restrict__ zxf, const float* __restrict__ zxb,
            const uint32_t* __restrict__ wtf, const uint32_t* __restrict__ wtb,
            const float* __restrict__ h0,  const float* __restrict__ c0,
            float* __restrict__ hf, float* __restrict__ hb) {
    const int dir = blockIdx.x;
    const float*    Z    = dir ? zxb : zxf;
    const uint32_t* WT   = dir ? wtb : wtf;   // (HD/2 x G4) packed pairs
    float*          hout = dir ? hb : hf;

    __shared__ float hsh[HD_DIM];
    __shared__ float gates[G4];

    const int j = threadIdx.x;
    float c = 0.f;
    if (j < HD_DIM) {
        hsh[j] = h0[dir * HD_DIM + j];
        c      = c0[dir * HD_DIM + j];
    }
    __syncthreads();

    const float2* h2 = (const float2*)hsh;

    for (int step = 0; step < S_LEN; ++step) {
        const int t = dir ? (S_LEN - 1 - step) : step;
        float acc = Z[(long)t * G4 + j];
        if (step + 1 < S_LEN) {
            const int tn = dir ? (t - 1) : (t + 1);
            __builtin_prefetch(&Z[(long)tn * G4 + j], 0, 0);  // global_prefetch_b8
        }
        const uint32_t* wcol = WT + j;
#pragma unroll 8
        for (int k2 = 0; k2 < HD_DIM / 2; ++k2) {
            const uint32_t w = wcol[(long)k2 * G4];           // coalesced 4B/lane
            const float2   h = h2[k2];                        // ds_load_b64 broadcast
            acc = fmaf(__uint_as_float(w << 16),         h.x, acc);
            acc = fmaf(__uint_as_float(w & 0xffff0000u), h.y, acc);
        }

        float act;
        if (j < 2 * HD_DIM || j >= 3 * HD_DIM)
            act = 1.0f / (1.0f + __expf(-acc));               // i, f, o gates
        else
            act = tanhf(acc);                                 // g gate
        gates[j] = act;
        __syncthreads();

        if (j < HD_DIM) {
            const float ig = gates[j];
            const float fg = gates[HD_DIM + j];
            const float gg = gates[2 * HD_DIM + j];
            const float og = gates[3 * HD_DIM + j];
            c = fmaf(fg, c, ig * gg);
            const float h = og * tanhf(c);
            hsh[j] = h;
            hout[(long)t * HD_DIM + j] = h;
        }
        __syncthreads();
    }
}

// ---------------- 5) feats = [hf|hb] @ W_out^T + b_out ---------------------
// grid (128 Mtiles, 3 Ntiles), block = 32
__global__ void k_gemm_feats(const float* __restrict__ hf,
                             const float* __restrict__ hb,
                             const bf16_t* __restrict__ wout,
                             const float* __restrict__ bout,
                             float* __restrict__ feats) {
    const int lane = threadIdx.x;
    const int mn   = lane & 15;
    const int hi   = lane >> 4;
    const int mt   = blockIdx.x;
    const int nt   = blockIdx.y;

    const int trow = mt * 16 + mn;
    const bf16_t* brow = wout + (long)(nt * 16 + mn) * H_DIM;

    union { v8f v; float s[8]; } acc;
    acc.v = (v8f){0.f,0.f,0.f,0.f,0.f,0.f,0.f,0.f};

    for (int k0 = 0; k0 < H_DIM; k0 += 32) {
        union { v16bf v; bf16_t s[16]; } a, b;
#pragma unroll
        for (int e = 0; e < 16; ++e) {
            const int k = k0 + wmma_kmap(e, hi);
            const float av = (k < HD_DIM) ? hf[(long)trow * HD_DIM + k]
                                          : hb[(long)trow * HD_DIM + (k - HD_DIM)];
            a.s[e] = (bf16_t)av;
            b.s[e] = brow[k];
        }
        acc.v = __builtin_amdgcn_wmma_f32_16x16x32_bf16(
            false, a.v, false, b.v, (short)0, acc.v, false, false);
    }

    const int col = nt * 16 + mn;
    const float bc = bout[col];
#pragma unroll
    for (int r = 0; r < 8; ++r) {
        const int row = mt * 16 + r + hi * 8;
        feats[(long)row * T_TAGS + col] = acc.s[r] + bc;
    }
}

// ---------------- 6) Viterbi (single workgroup) ----------------------------
__global__ void k_viterbi(const float* __restrict__ feats,
                          const float* __restrict__ trans,
                          int* __restrict__ bps,
                          float* __restrict__ path) {
    __shared__ float prev[T_TAGS];
    __shared__ float tr[T_TAGS * T_TAGS];
    const int j = threadIdx.x;

    for (int i = j; i < T_TAGS * T_TAGS; i += blockDim.x) tr[i] = trans[i];
    if (j < T_TAGS) prev[j] = feats[j];
    __syncthreads();

    for (int t = 1; t < S_LEN; ++t) {
        float best = -3.4e38f;
        int bi = 0;
        if (j < T_TAGS) {
            for (int i = 0; i < T_TAGS; ++i) {
                const float sc = prev[i] + tr[i * T_TAGS + j];
                if (sc > best) { best = sc; bi = i; }   // first-max argmax semantics
            }
            best += feats[(long)t * T_TAGS + j];
        }
        __syncthreads();
        if (j < T_TAGS) {
            prev[j] = best;
            bps[(long)(t - 1) * T_TAGS + j] = bi;
        }
        __syncthreads();
    }

    if (j == 0) {
        int cur = 0;
        float best = prev[0];
        for (int i = 1; i < T_TAGS; ++i)
            if (prev[i] > best) { best = prev[i]; cur = i; }
        path[S_LEN - 1] = (float)cur;
        for (int t = S_LEN - 2; t >= 0; --t) {
            cur = bps[(long)t * T_TAGS + cur];
            path[t] = (float)cur;
        }
    }
}

// ---------------------------------------------------------------------------
extern "C" void kernel_launch(void* const* d_in, const int* in_sizes, int n_in,
                              void* d_out, int out_size, void* d_ws, size_t ws_size,
                              hipStream_t stream) {
    const int*   sent  = (const int*)  d_in[0];
    const float* table = (const float*)d_in[1];
    const float* Wih_f = (const float*)d_in[2];
    const float* Whh_f = (const float*)d_in[3];
    const float* b_f   = (const float*)d_in[4];
    const float* Wih_b = (const float*)d_in[5];
    const float* Whh_b = (const float*)d_in[6];
    const float* b_b   = (const float*)d_in[7];
    const float* W_out = (const float*)d_in[8];
    const float* b_out = (const float*)d_in[9];
    const float* trans = (const float*)d_in[10];
    const float* h0    = (const float*)d_in[11];
    const float* c0    = (const float*)d_in[12];
    (void)in_sizes; (void)n_in; (void)out_size; (void)ws_size;

    char* ws = (char*)d_ws;
    bf16_t*   xbf   = (bf16_t*)  (ws + OFF_XBF);
    bf16_t*   wihf  = (bf16_t*)  (ws + OFF_WIHF);
    bf16_t*   wihb  = (bf16_t*)  (ws + OFF_WIHB);
    bf16_t*   woutb = (bf16_t*)  (ws + OFF_WOUTBF);
    uint32_t* whhtf = (uint32_t*)(ws + OFF_WHHTF);
    uint32_t* whhtb = (uint32_t*)(ws + OFF_WHHTB);
    float*    zxf   = (float*)   (ws + OFF_ZXF);
    float*    zxb   = (float*)   (ws + OFF_ZXB);
    float*    hf    = (float*)   (ws + OFF_HF);
    float*    hb    = (float*)   (ws + OFF_HB);
    float*    feats = (float*)   (ws + OFF_FEATS);
    int*      bps   = (int*)     (ws + OFF_BPS);

    // 1) embedding gather
    k_embed<<<S_LEN, 256, 0, stream>>>(sent, table, xbf);

    // 2) weight prep
    k_cvt_bf16<<<(G4 * E_DIM + 255) / 256, 256, 0, stream>>>(Wih_f, wihf, G4 * E_DIM);
    k_cvt_bf16<<<(G4 * E_DIM + 255) / 256, 256, 0, stream>>>(Wih_b, wihb, G4 * E_DIM);
    k_cvt_bf16<<<(T_TAGS * H_DIM + 255) / 256, 256, 0, stream>>>(W_out, woutb, T_TAGS * H_DIM);
    k_transpose_pack<<<((HD_DIM / 2) * G4 + 255) / 256, 256, 0, stream>>>(Whh_f, whhtf);
    k_transpose_pack<<<((HD_DIM / 2) * G4 + 255) / 256, 256, 0, stream>>>(Whh_b, whhtb);

    // 3) batched input projection (TDM staging + WMMA)
    k_gemm_zx<<<dim3(S_LEN / 16, G4 / 16, 2), 32, 0, stream>>>(
        xbf, wihf, wihb, b_f, b_b, zxf, zxb);

    // 4) recurrence (both directions concurrently)
    k_lstm<<<2, 1024, 0, stream>>>(zxf, zxb, whhtf, whhtb, h0, c0, hf, hb);

    // 5) emission scores (WMMA)
    k_gemm_feats<<<dim3(S_LEN / 16, T_TAGS / 16), 32, 0, stream>>>(
        hf, hb, woutb, b_out, feats);

    // 6) Viterbi decode
    k_viterbi<<<1, 64, 0, stream>>>(feats, trans, bps, (float*)d_out);
}
